// VolumeModel_52527450030770
// MI455X (gfx1250) — matrix operand order repaired and checked by reading, not compiled, not run on hardware
//
#include <hip/hip_runtime.h>
#include <hip/hip_bf16.h>

typedef __attribute__((ext_vector_type(16))) _Float16     v16h;
typedef __attribute__((ext_vector_type(8)))  float        v8f;
typedef __attribute__((ext_vector_type(4)))  unsigned int u32x4;
typedef __attribute__((ext_vector_type(8)))  int          i32x8;
typedef __attribute__((ext_vector_type(4)))  int          i32x4;

#define IMG   64
#define NPTS  300
#define GRIDD 16
#define BATCH 8
#define NLAT  16384     // 4 * 16^3
#define KDIM  480
#define EPSW  1e-10f

#if __has_builtin(__builtin_amdgcn_tensor_load_to_lds) && \
    __has_builtin(__builtin_amdgcn_s_wait_tensorcnt)
#define HAVE_TDM 1
#else
#define HAVE_TDM 0
#endif

__device__ __forceinline__ float sigmoidf(float x) {
  return 1.0f / (1.0f + __expf(-x));
}

// ---------------------------------------------------------------------------
// Kernel 1: layers 1 & 2 of the MLP. Tiny (8x60, 8x480). One block.
// ---------------------------------------------------------------------------
__global__ __launch_bounds__(256)
void mlp12_kernel(const float* __restrict__ zs,
                  const float* __restrict__ W1, const float* __restrict__ b1,
                  const float* __restrict__ W2, const float* __restrict__ b2,
                  float* __restrict__ h2out) {
  __shared__ float h1[BATCH * 60];
  const int tid = threadIdx.x;

  for (int idx = tid; idx < BATCH * 60; idx += 256) {
    const int i = idx / 60, j = idx % 60;
    float acc = b1[j];
    #pragma unroll 5
    for (int k = 0; k < 30; ++k) acc += zs[i * 30 + k] * W1[k * 60 + j];
    h1[idx] = sigmoidf(acc);
  }
  __syncthreads();

  for (int idx = tid; idx < BATCH * KDIM; idx += 256) {
    const int i = idx / KDIM, j = idx % KDIM;
    float acc = b2[j];
    #pragma unroll 5
    for (int k = 0; k < 60; ++k) acc += h1[i * 60 + k] * W2[k * KDIM + j];
    h2out[idx] = sigmoidf(acc);
  }
}

// ---------------------------------------------------------------------------
// Kernel 2: layer 3 via WMMA. D = sigmoid(h2[8x480] @ W3[480x16384] + b3).
// M padded 8->16, K = 15 tiles of 32, one wave per 16-wide N tile (1024 waves).
// W3 (31.5 MB — the roofline-dominant tensor) is streamed from HBM exactly
// once: each 16-column slab is owned by exactly one wave.
// ---------------------------------------------------------------------------
__global__ __launch_bounds__(256)
void layer3_wmma_kernel(const float* __restrict__ h2,
                        const float* __restrict__ W3,
                        const float* __restrict__ b3,
                        float* __restrict__ lat) {
  const int wave = (blockIdx.x * blockDim.x + threadIdx.x) >> 5;  // N tile id
  if (wave >= NLAT / 16) return;
  const int lane = threadIdx.x & 31;
  const int m    = lane & 15;          // A row (lo+hi lanes), also B/D column
  const bool hi  = lane >= 16;
  const int abase = hi ? 16 : 0;       // A: hi lanes hold K 16..31
  const int kb    = hi ? 16 : 0;       // B: hi lanes hold K 16..31
  const int col0  = wave * 16;

  v8f acc = {0.f, 0.f, 0.f, 0.f, 0.f, 0.f, 0.f, 0.f};

  #pragma unroll 5
  for (int kt = 0; kt < KDIM / 32; ++kt) {
    const int k0 = kt * 32;

    // A fragment: 16x32 fp16, rows 8..15 are zero padding.
    v16h a;
    #pragma unroll
    for (int h = 0; h < 16; ++h) a[h] = (_Float16)0.f;
    if (m < BATCH) {
      const float* hrow = h2 + m * KDIM + k0;
      #pragma unroll
      for (int h = 0; h < 8; ++h) {
        a[h]     = (_Float16)hrow[abase + h];        // K = k0 + abase + h
        a[8 + h] = (_Float16)hrow[abase + 8 + h];    // K = k0 + abase + 8 + h
      }
    }

    // B fragment: 32x16 fp16, lane = column, halfs = consecutive K.
    v16h b;
    const float* wcol = W3 + (size_t)(k0 + kb) * NLAT + (col0 + m);
    #pragma unroll
    for (int h = 0; h < 16; ++h) b[h] = (_Float16)wcol[(size_t)h * NLAT];

    acc = __builtin_amdgcn_wmma_f32_16x16x32_f16(
        /*neg_a=*/false, a, /*neg_b=*/false, b,
        /*c_mod=*/(short)0, acc, /*reuse_a=*/false, /*reuse_b=*/false);
  }

  // D layout: vgpr r, lanes 0-15 -> row r, col lane; lanes 16-31 -> row r+8.
  // Valid rows are 0..7 => entirely in the lo lanes.
  if (!hi) {
    const int col = col0 + m;
    const float bias = b3[col];
    #pragma unroll
    for (int r = 0; r < BATCH; ++r) {
      lat[(size_t)r * NLAT + col] = sigmoidf(acc[r] + bias);
    }
  }
}

// ---------------------------------------------------------------------------
// Kernel 3: ray march. One block per (256-ray tile, batch). The batch's whole
// 4x16^3 volume (64 KB) is DMA'd into LDS by the Tensor Data Mover (one wave
// issues the descriptor; TENSORcnt + workgroup barrier make it visible),
// then one thread marches each ray (300 steps, border-zero trilinear taps,
// sequential compositing). 64 KB is a fifth of the WGP's 320 KB LDS.
// ---------------------------------------------------------------------------
__device__ __forceinline__ void sample4(const float* vol,
                                        float lx, float ly, float lz,
                                        float& sig, float& c0, float& c1, float& c2) {
  const float ix = (lx + 1.0f) * 7.5f;
  const float iy = (ly + 1.0f) * 7.5f;
  const float iz = (lz + 1.0f) * 7.5f;
  const float xf = floorf(ix), yf = floorf(iy), zf = floorf(iz);
  const float tx = ix - xf, ty = iy - yf, tz = iz - zf;
  sig = 0.f; c0 = 0.f; c1 = 0.f; c2 = 0.f;
  #pragma unroll
  for (int dx = 0; dx < 2; ++dx) {
    const float xc = xf + dx;
    const float wx = dx ? tx : (1.0f - tx);
    const bool vx = (xc >= 0.0f) && (xc <= 15.0f);
    const int xi = (int)fminf(fmaxf(xc, 0.0f), 15.0f);
    #pragma unroll
    for (int dy = 0; dy < 2; ++dy) {
      const float yc = yf + dy;
      const float wy = dy ? ty : (1.0f - ty);
      const bool vy = (yc >= 0.0f) && (yc <= 15.0f);
      const int yi = (int)fminf(fmaxf(yc, 0.0f), 15.0f);
      #pragma unroll
      for (int dz = 0; dz < 2; ++dz) {
        const float zc = zf + dz;
        const float wz = dz ? tz : (1.0f - tz);
        const bool vz = (zc >= 0.0f) && (zc <= 15.0f);
        const int zi = (int)fminf(fmaxf(zc, 0.0f), 15.0f);
        const float w = (vx && vy && vz) ? (wx * wy * wz) : 0.0f;
        const int off = zi * 256 + yi * 16 + xi;
        sig += w * vol[off];
        c0  += w * vol[4096  + off];
        c1  += w * vol[8192  + off];
        c2  += w * vol[12288 + off];
      }
    }
  }
}

__global__ __launch_bounds__(256)
void render_kernel(const float* __restrict__ lat,
                   const float* __restrict__ R,
                   const float* __restrict__ T,
                   float* __restrict__ out_opac,   // [B,64,64,1]
                   float* __restrict__ out_rgb) {  // [B,64,64,3]
  __shared__ float vol[4 * 4096];  // 64 KB; sole LDS alloc -> LDS offset 0
  const int b = blockIdx.y;
  const float* src = lat + (size_t)b * NLAT;

#if HAVE_TDM
  if (threadIdx.x < 32) {
    // Tensor DMA descriptor (ISA 8.3/8.4): 1D copy of 16384 dwords to LDS 0.
    const unsigned long long ga = (unsigned long long)(const void*)src;
    u32x4 g0 = {
        1u,                                               // count=1, user mode
        0u,                                               // lds_addr = 0 (vol)
        (unsigned int)(ga & 0xFFFFFFFFu),                 // global_addr[31:0]
        (unsigned int)((ga >> 32) & 0x01FFFFFFu) | (2u << 30)  // addr[56:32]|type=2
    };
    i32x8 g1 = {
        0x00020000,   // workgroup_mask=0, data_size=2 (4 bytes)
        0x40000000,   // bar_addr=0, tensor_dim0[15:0]=16384
        0x00010000,   // tensor_dim0[31:16]=0, tensor_dim1=1
        0x40000000,   // tensor_dim1[31:16]=0, tile_dim0=16384
        0x00000001,   // tile_dim1=1, tile_dim2=0
        0x00004000,   // tensor_dim0_stride[31:0]=16384
        0, 0          // stride hi / dim1_stride = 0
    };
    i32x4 gz = {0, 0, 0, 0};
#if defined(__clang_major__) && (__clang_major__ >= 23)
    i32x8 gz8 = {0, 0, 0, 0, 0, 0, 0, 0};
    __builtin_amdgcn_tensor_load_to_lds(g0, g1, gz, gz, gz8, 0);
#else
    __builtin_amdgcn_tensor_load_to_lds(g0, g1, gz, gz, 0);
#endif
    __builtin_amdgcn_s_wait_tensorcnt(0);
  }
  // The TDM builtin carries no pointer operand, so alias analysis cannot see
  // that it writes `vol`. Escape the array's address into an opaque asm with
  // a memory clobber: the optimizer must now assume `vol` was written, which
  // keeps the LDS sampler below alive. (A bare ::: "memory" clobber is NOT
  // enough — a non-escaping local LDS object is provably unreachable from it.)
  {
    float* volp = vol;
    asm volatile("" : "+v"(volp) :: "memory");
  }
  __syncthreads();
#else
  for (int i = threadIdx.x; i < NLAT; i += 256) vol[i] = src[i];
  __syncthreads();
#endif

  const int ray = blockIdx.x * 256 + threadIdx.x;  // 0..4095
  const int h = ray >> 6;
  const int w = ray & 63;

  // Camera: dirs_w[d] = sum_c dir_cam[c] * R[d][c]; origin[d] = -sum_c T[c]*R[d][c]
  float Rm[9], Tb[3];
  #pragma unroll
  for (int i = 0; i < 9; ++i) Rm[i] = R[b * 9 + i];
  #pragma unroll
  for (int i = 0; i < 3; ++i) Tb[i] = T[b * 3 + i];

  const float invf = 1.0f / 1.73205080757f;            // tan(30 deg)
  const float cx = ((63.0f - 2.0f * (float)w) / 64.0f) * invf;
  const float cy = ((63.0f - 2.0f * (float)h) / 64.0f) * invf;

  float dir[3], org[3];
  #pragma unroll
  for (int d = 0; d < 3; ++d) {
    dir[d] = cx * Rm[d * 3 + 0] + cy * Rm[d * 3 + 1] + Rm[d * 3 + 2];
    org[d] = -(Tb[0] * Rm[d * 3 + 0] + Tb[1] * Rm[d * 3 + 1] + Tb[2] * Rm[d * 3 + 2]);
  }

  const float inv_he = 1.0f / 0.75f;                   // half_extent = 15*0.1/2
  const float dstep = 2.9f / 299.0f;

  float trans  = 1.0f;  // cumprod of (1+eps - sigma), exclusive
  float transo = 1.0f;  // prod of (1 - sigma)
  float cr = 0.f, cg = 0.f, cb = 0.f;

  for (int s = 0; s < NPTS; ++s) {
    const float depth = 0.1f + dstep * (float)s;
    const float lx = (org[0] + dir[0] * depth) * inv_he;
    const float ly = (org[1] + dir[1] * depth) * inv_he;
    const float lz = (org[2] + dir[2] * depth) * inv_he;
    float sig, c0, c1, c2;
    sample4(vol, lx, ly, lz, sig, c0, c1, c2);
    const float wgt = sig * trans;
    cr += wgt * c0; cg += wgt * c1; cb += wgt * c2;
    trans  *= (1.0f + EPSW - sig);
    transo *= (1.0f - sig);
  }

  const int pix = b * (IMG * IMG) + ray;
  out_opac[pix] = 1.0f - transo;
  out_rgb[pix * 3 + 0] = cr;
  out_rgb[pix * 3 + 1] = cg;
  out_rgb[pix * 3 + 2] = cb;
}

// ---------------------------------------------------------------------------
extern "C" void kernel_launch(void* const* d_in, const int* in_sizes, int n_in,
                              void* d_out, int out_size, void* d_ws, size_t ws_size,
                              hipStream_t stream) {
  (void)in_sizes; (void)n_in; (void)out_size; (void)ws_size;
  const float* zs = (const float*)d_in[0];
  const float* W1 = (const float*)d_in[1];
  const float* b1 = (const float*)d_in[2];
  const float* W2 = (const float*)d_in[3];
  const float* b2 = (const float*)d_in[4];
  const float* W3 = (const float*)d_in[5];
  const float* b3 = (const float*)d_in[6];
  const float* R  = (const float*)d_in[7];
  const float* T  = (const float*)d_in[8];

  float* h2  = (float*)d_ws;            // 8*480 floats
  float* lat = (float*)d_ws + 4096;     // 8*16384 floats

  float* out_opac = (float*)d_out;                       // 8*64*64
  float* out_rgb  = (float*)d_out + BATCH * IMG * IMG;   // 8*64*64*3

  mlp12_kernel<<<1, 256, 0, stream>>>(zs, W1, b1, W2, b2, h2);

  // 1024 N-tiles of 16 columns, 8 waves (256 threads) per block -> 128 blocks
  layer3_wmma_kernel<<<128, 256, 0, stream>>>(h2, W3, b3, lat);

  dim3 rgrid(IMG * IMG / 256, BATCH);   // (16, 8)
  render_kernel<<<rgrid, 256, 0, stream>>>(lat, R, T, out_opac, out_rgb);
}
